// FeaturePropagation_64622077935822
// MI455X (gfx1250) — compile-verified, matrix operand order checked by reading, not run on hardware
//
#include <hip/hip_runtime.h>
#include <hip/hip_bf16.h>
#include <cstdint>
#include <cstddef>

// ---------------------------------------------------------------------------
// FeaturePropagation (PointNet++ FP module) for MI455X / gfx1250, wave32.
// GEMMs run on v_wmma_f32_16x16x32_bf16 (f32 accumulate), software-pipelined:
// fragments for k-step s+1 are prefetched before the WMMAs of step s so the
// per-wave s_wait_loadcnt becomes partial instead of 0.
// Wave tile: 32(o) x 64(n) = 8 accumulators; block = 8 waves = all 256 o.
// ---------------------------------------------------------------------------

typedef __attribute__((ext_vector_type(16))) __bf16 v16bf;
typedef __attribute__((ext_vector_type(8)))  __bf16 v8bf;
typedef __attribute__((ext_vector_type(8)))  float  v8f;

#define B_   8
#define N1_  16384
#define N2_  1024
#define C1_  128
#define C2_  256
#define CF_  384   // C1 + C2
#define OC_  256   // output channels of both layers

static __device__ __forceinline__ __bf16 f2bf(float f) {
    union { float f; uint32_t u; } v; v.f = f;
    uint32_t u = v.u;
    uint32_t r = u + 0x7FFFu + ((u >> 16) & 1u);   // round-to-nearest-even
    uint16_t h = (uint16_t)(r >> 16);
    __bf16 out;
    __builtin_memcpy(&out, &h, 2);
    return out;
}

// ISA 16-bit A/B fragment (16x32 / 32x16): lane l owns row/col (l&15);
// elements [0..7] cover K = (l>>4)*8 .. +7 (contiguous 16B), elements
// [8..15] cover K = 16+(l>>4)*8 .. +7 (contiguous 16B). Row-major K-contig
// storage -> two aligned b128 loads per lane.
static __device__ __forceinline__ v16bf load_frag(const __bf16* __restrict__ row,
                                                  int k0, int half) {
    v16bf f;
    const v8bf lo = *(const v8bf*)(row + k0 + half * 8);
    const v8bf hi = *(const v8bf*)(row + k0 + 16 + half * 8);
    #pragma unroll
    for (int i = 0; i < 8; ++i) { f[i] = lo[i]; f[8 + i] = hi[i]; }
    return f;
}

// --------------------------- weight conversion -----------------------------
__global__ void cvt_bf16_kernel(const float* __restrict__ src,
                                __bf16* __restrict__ dst, int n) {
    int i = blockIdx.x * blockDim.x + threadIdx.x;
    if (i < n) dst[i] = f2bf(src[i]);
}

// ------------------------------- 3-NN --------------------------------------
__global__ __launch_bounds__(256)
void knn3_kernel(const float* __restrict__ xyz1, const float* __restrict__ xyz2,
                 int* __restrict__ idx, float* __restrict__ wgt) {
    __shared__ float sx[256], sy[256], sz[256];
    const int n = blockIdx.x * 256 + threadIdx.x;
    const int b = blockIdx.y;
    const size_t pbase = ((size_t)b * N1_ + n) * 3;
    const float px = xyz1[pbase + 0];
    const float py = xyz1[pbase + 1];
    const float pz = xyz1[pbase + 2];

    float d0 = 3.4e38f, d1 = 3.4e38f, d2 = 3.4e38f;
    int   i0 = 0, i1 = 0, i2 = 0;

    for (int m0 = 0; m0 < N2_; m0 += 256) {
        __syncthreads();
        const size_t sbase = ((size_t)b * N2_ + m0 + threadIdx.x) * 3;
        sx[threadIdx.x] = xyz2[sbase + 0];
        sy[threadIdx.x] = xyz2[sbase + 1];
        sz[threadIdx.x] = xyz2[sbase + 2];
        __syncthreads();
        #pragma unroll 8
        for (int j = 0; j < 256; ++j) {
            const float dx = px - sx[j], dy = py - sy[j], dz = pz - sz[j];
            const float d = dx * dx + dy * dy + dz * dz;
            const int m = m0 + j;
            if (d < d0)      { d2 = d1; i2 = i1; d1 = d0; i1 = i0; d0 = d; i0 = m; }
            else if (d < d1) { d2 = d1; i2 = i1; d1 = d;  i1 = m; }
            else if (d < d2) { d2 = d;  i2 = m; }
        }
    }

    float w0 = 1.0f / (sqrtf(fmaxf(d0, 0.0f)) + 1e-8f);
    float w1 = 1.0f / (sqrtf(fmaxf(d1, 0.0f)) + 1e-8f);
    float w2 = 1.0f / (sqrtf(fmaxf(d2, 0.0f)) + 1e-8f);
    const float s = 1.0f / (w0 + w1 + w2);
    const size_t o = ((size_t)b * N1_ + n) * 3;
    idx[o + 0] = i0; idx[o + 1] = i1; idx[o + 2] = i2;
    wgt[o + 0] = w0 * s; wgt[o + 1] = w1 * s; wgt[o + 2] = w2 * s;
}

// ---------------- build fused [B][N1][384] bf16 (n-major) ------------------
__global__ __launch_bounds__(128)
void build_fused_kernel(const float* __restrict__ f1, const float* __restrict__ f2,
                        const int* __restrict__ idx, const float* __restrict__ wgt,
                        __bf16* __restrict__ fused) {
    const int c = blockIdx.x * 128 + threadIdx.x;   // 0..383
    const int n = blockIdx.y;
    const int b = blockIdx.z;
    float v;
    if (c < C1_) {
        v = f1[((size_t)b * C1_ + c) * N1_ + n];
    } else {
        const int c2 = c - C1_;
        const size_t base = ((size_t)b * C2_ + c2) * N2_;
        const size_t t = ((size_t)b * N1_ + n) * 3;
        v = wgt[t + 0] * f2[base + idx[t + 0]]
          + wgt[t + 1] * f2[base + idx[t + 1]]
          + wgt[t + 2] * f2[base + idx[t + 2]];
    }
    fused[((size_t)b * N1_ + n) * CF_ + c] = f2bf(v);
}

// ------------------------------ WMMA GEMM ----------------------------------
// D[OC x N1] = A[OC x K] * Bm^T, Bm stored n-major [B*N1][K].
// Block = 8 waves; wave w computes o in [w*32, w*32+32) x 64 n-columns.
// Register double-buffered over the K loop.
template <int K, bool OMAJOR_OUT>
__global__ __launch_bounds__(256)
void wmma_gemm_kernel(const __bf16* __restrict__ A,
                      const __bf16* __restrict__ Bm,
                      float* __restrict__ out) {
    const int lane = threadIdx.x & 31;
    const int wave = threadIdx.x >> 5;
    const int half = lane >> 4;
    const int l15  = lane & 15;
    const int o_base = wave * 32;          // 8 waves cover all 256 channels
    const int n0 = blockIdx.x * 64;
    const int b  = blockIdx.y;

    const __bf16* Arow0 = A + (size_t)(o_base + l15) * K;
    const __bf16* Arow1 = Arow0 + (size_t)16 * K;
    const __bf16* Brow[4];
    #pragma unroll
    for (int t = 0; t < 4; ++t)
        Brow[t] = Bm + ((size_t)b * N1_ + n0 + t * 16 + l15) * K;

    v8f acc[2][4];
    #pragma unroll
    for (int i = 0; i < 2; ++i)
        #pragma unroll
        for (int t = 0; t < 4; ++t) acc[i][t] = v8f{};

    constexpr int NS = K / 32;

    // prologue: fragments for k-step 0
    v16bf a0 = load_frag(Arow0, 0, half);
    v16bf a1 = load_frag(Arow1, 0, half);
    v16bf bb[4];
    #pragma unroll
    for (int t = 0; t < 4; ++t) bb[t] = load_frag(Brow[t], 0, half);

    #pragma unroll
    for (int s = 0; s < NS; ++s) {
        // prefetch next k-step while current WMMAs execute
        v16bf a0n = a0, a1n = a1, bbn[4];
        #pragma unroll
        for (int t = 0; t < 4; ++t) bbn[t] = bb[t];
        if (s + 1 < NS) {
            const int k0 = (s + 1) * 32;
            a0n = load_frag(Arow0, k0, half);
            a1n = load_frag(Arow1, k0, half);
            #pragma unroll
            for (int t = 0; t < 4; ++t) bbn[t] = load_frag(Brow[t], k0, half);
        }
        #pragma unroll
        for (int t = 0; t < 4; ++t) {
            acc[0][t] = __builtin_amdgcn_wmma_f32_16x16x32_bf16(
                false, a0, false, bb[t], (short)0, acc[0][t], false, false);
            acc[1][t] = __builtin_amdgcn_wmma_f32_16x16x32_bf16(
                false, a1, false, bb[t], (short)0, acc[1][t], false, false);
        }
        a0 = a0n; a1 = a1n;
        #pragma unroll
        for (int t = 0; t < 4; ++t) bb[t] = bbn[t];
    }

    #pragma unroll
    for (int t = 0; t < 4; ++t) {
        const int n = n0 + t * 16 + l15;
        #pragma unroll
        for (int i = 0; i < 2; ++i) {
            if constexpr (!OMAJOR_OUT) {
                // n-major [B*N1][OC]: 8 accum floats contiguous in o
                *(v8f*)(out + ((size_t)b * N1_ + n) * OC_ +
                        o_base + i * 16 + 8 * half) = acc[i][t];
            } else {
                // o-major [B][OC][N1]: per-VGPR stores form 2x64B segments
                #pragma unroll
                for (int r = 0; r < 8; ++r) {
                    const int o = o_base + i * 16 + 8 * half + r;
                    out[((size_t)b * OC_ + o) * N1_ + n] = acc[i][t][r];
                }
            }
        }
    }
}

// ----------------- deterministic BN statistics reductions ------------------
__global__ __launch_bounds__(256)
void reduce_nmajor_kernel(const float* __restrict__ x,
                          float* __restrict__ psum, float* __restrict__ psq) {
    const int o = threadIdx.x;
    const size_t r0 = (size_t)blockIdx.x * 512;
    float s = 0.0f, q = 0.0f;
    for (int r = 0; r < 512; ++r) {
        const float v = x[(r0 + r) * OC_ + o];
        s += v; q += v * v;
    }
    psum[(size_t)blockIdx.x * OC_ + o] = s;
    psq [(size_t)blockIdx.x * OC_ + o] = q;
}

__global__ __launch_bounds__(256)
void reduce_omajor_kernel(const float* __restrict__ x,
                          float* __restrict__ psum, float* __restrict__ psq) {
    __shared__ float ss[256], sq[256];
    const int o = blockIdx.x, b = blockIdx.y;
    const float* row = x + ((size_t)b * OC_ + o) * N1_;
    float s = 0.0f, q = 0.0f;
    for (int n = threadIdx.x; n < N1_; n += 256) {
        const float v = row[n];
        s += v; q += v * v;
    }
    ss[threadIdx.x] = s; sq[threadIdx.x] = q;
    __syncthreads();
    for (int st = 128; st > 0; st >>= 1) {
        if (threadIdx.x < st) {
            ss[threadIdx.x] += ss[threadIdx.x + st];
            sq[threadIdx.x] += sq[threadIdx.x + st];
        }
        __syncthreads();
    }
    if (threadIdx.x == 0) {
        psum[(size_t)b * OC_ + o] = ss[0];
        psq [(size_t)b * OC_ + o] = sq[0];
    }
}

__global__ void finalize_stats_kernel(const float* __restrict__ psum,
                                      const float* __restrict__ psq,
                                      int nparts, float* __restrict__ stats) {
    const int o = blockIdx.x * blockDim.x + threadIdx.x;
    if (o >= OC_) return;
    float s = 0.0f, q = 0.0f;
    for (int p = 0; p < nparts; ++p) {
        s += psum[(size_t)p * OC_ + o];
        q += psq [(size_t)p * OC_ + o];
    }
    const float inv_cnt = 1.0f / (float)((size_t)B_ * N1_);
    const float mean = s * inv_cnt;
    const float var  = q * inv_cnt - mean * mean;
    stats[o]        = mean;
    stats[OC_ + o]  = rsqrtf(var + 1e-5f);
}

// --------------------------- BN + ReLU apply -------------------------------
__global__ __launch_bounds__(256)
void bn_relu_to_bf16_kernel(const float* __restrict__ x, const float* __restrict__ stats,
                            const float* __restrict__ g, const float* __restrict__ bta,
                            __bf16* __restrict__ y) {
    const size_t e = (size_t)blockIdx.x * 256 + threadIdx.x;
    const int o = threadIdx.x;                 // row length == blockDim == 256
    float v = x[e];
    v = (v - stats[o]) * stats[OC_ + o] * g[o] + bta[o];
    y[e] = f2bf(fmaxf(v, 0.0f));
}

__global__ __launch_bounds__(256)
void bn_relu_final_kernel(const float* __restrict__ x, const float* __restrict__ stats,
                          const float* __restrict__ g, const float* __restrict__ bta,
                          float* __restrict__ out) {
    const int n = blockIdx.x * 256 + threadIdx.x;
    const int o = blockIdx.y, b = blockIdx.z;
    const size_t e = ((size_t)b * OC_ + o) * N1_ + n;
    float v = x[e];
    v = (v - stats[o]) * stats[OC_ + o] * g[o] + bta[o];
    out[e] = fmaxf(v, 0.0f);
}

// ---------------------------------------------------------------------------
extern "C" void kernel_launch(void* const* d_in, const int* in_sizes, int n_in,
                              void* d_out, int out_size, void* d_ws, size_t ws_size,
                              hipStream_t stream) {
    const float* xyz1 = (const float*)d_in[0];
    const float* xyz2 = (const float*)d_in[1];
    const float* f1   = (const float*)d_in[2];
    const float* f2   = (const float*)d_in[3];
    const float* w1   = (const float*)d_in[4];
    const float* g1   = (const float*)d_in[5];
    const float* b1   = (const float*)d_in[6];
    const float* w2   = (const float*)d_in[7];
    const float* g2   = (const float*)d_in[8];
    const float* b2   = (const float*)d_in[9];
    float* out = (float*)d_out;

    // ---- workspace carve-up (256B aligned) ----
    uint8_t* ws = (uint8_t*)d_ws;
    size_t off = 0;
    auto alloc = [&](size_t bytes) -> uint8_t* {
        uint8_t* p = ws + off;
        off = (off + bytes + 255) & ~(size_t)255;
        return p;
    };
    int*    idx   = (int*)   alloc((size_t)B_ * N1_ * 3 * 4);
    float*  wgt   = (float*) alloc((size_t)B_ * N1_ * 3 * 4);
    __bf16* w1bf  = (__bf16*)alloc((size_t)OC_ * CF_ * 2);
    __bf16* w2bf  = (__bf16*)alloc((size_t)OC_ * OC_ * 2);
    __bf16* fused = (__bf16*)alloc((size_t)B_ * N1_ * CF_ * 2);
    float*  hpre  = (float*) alloc((size_t)B_ * N1_ * OC_ * 4);  // GEMM1 out (n-major)
    float*  outpre = hpre;                                        // aliased: GEMM2 out (o-major)
    __bf16* hbf   = (__bf16*)alloc((size_t)B_ * N1_ * OC_ * 2);
    float*  part1s = (float*)alloc((size_t)256 * OC_ * 4);
    float*  part1q = (float*)alloc((size_t)256 * OC_ * 4);
    float*  part2s = (float*)alloc((size_t)B_  * OC_ * 4);
    float*  part2q = (float*)alloc((size_t)B_  * OC_ * 4);
    float*  stats1 = (float*)alloc((size_t)2 * OC_ * 4);
    float*  stats2 = (float*)alloc((size_t)2 * OC_ * 4);
    (void)ws_size; (void)n_in; (void)in_sizes; (void)out_size;

    // 1) weights -> bf16
    cvt_bf16_kernel<<<dim3((OC_ * CF_ + 255) / 256), 256, 0, stream>>>(w1, w1bf, OC_ * CF_);
    cvt_bf16_kernel<<<dim3((OC_ * OC_ + 255) / 256), 256, 0, stream>>>(w2, w2bf, OC_ * OC_);

    // 2) 3-NN indices + normalized inverse-distance weights
    knn3_kernel<<<dim3(N1_ / 256, B_), 256, 0, stream>>>(xyz1, xyz2, idx, wgt);

    // 3) fused activations, bf16 n-major
    build_fused_kernel<<<dim3(CF_ / 128, N1_, B_), 128, 0, stream>>>(f1, f2, idx, wgt, fused);

    // 4) GEMM1: [256 x 384] x [384 x N1] -> hpre (n-major)
    wmma_gemm_kernel<CF_, false><<<dim3(N1_ / 64, B_), 256, 0, stream>>>(w1bf, fused, hpre);

    // 5) BN1 stats (deterministic two-stage) + apply + bf16
    reduce_nmajor_kernel<<<dim3((B_ * N1_) / 512), 256, 0, stream>>>(hpre, part1s, part1q);
    finalize_stats_kernel<<<dim3(1), 256, 0, stream>>>(part1s, part1q, 256, stats1);
    bn_relu_to_bf16_kernel<<<dim3(B_ * N1_), 256, 0, stream>>>(hpre, stats1, g1, b1, hbf);

    // 6) GEMM2: [256 x 256] x [256 x N1] -> outpre (o-major, aliases hpre)
    wmma_gemm_kernel<OC_, true><<<dim3(N1_ / 64, B_), 256, 0, stream>>>(w2bf, hbf, outpre);

    // 7) BN2 stats + final apply into d_out ([B, 256, N1] f32)
    reduce_omajor_kernel<<<dim3(OC_, B_), 256, 0, stream>>>(outpre, part2s, part2q);
    finalize_stats_kernel<<<dim3(1), 256, 0, stream>>>(part2s, part2q, B_, stats2);
    bn_relu_final_kernel<<<dim3(N1_ / 256, OC_, B_), 256, 0, stream>>>(outpre, stats2, g2, b2, out);
}
